// BinaryLinear_9552007266663
// MI455X (gfx1250) — compile-verified, matrix operand order checked by reading, not run on hardware
//
#include <hip/hip_runtime.h>
#include <hip/hip_bf16.h>

#define B_DIM   8192
#define IN_DIM  4096
#define OUT_DIM 4096
#define BM 128
#define BN 256
#define BK 32
#define LDSS 40   // padded LDS row stride in bf16 elements (80B) -> conflict-free b128 frag loads
#define EPS 1e-5f

typedef __attribute__((ext_vector_type(16))) __bf16 v16bf;
typedef __attribute__((ext_vector_type(8)))  float  v8f;
typedef __attribute__((ext_vector_type(4)))  int    v4i;

#if defined(__gfx1250__) && __has_builtin(__builtin_amdgcn_global_load_async_to_lds_b128)
#define HAVE_ASYNC_LDS 1
typedef __attribute__((address_space(1))) v4i gv4i;  // global 128-bit payload
typedef __attribute__((address_space(3))) v4i lv4i;  // LDS 128-bit payload
#endif

// ---------- helpers ----------
__device__ inline unsigned short f2bf_rne(float f) {
    unsigned int u = __float_as_uint(f);
    unsigned int r = u + 0x7FFFu + ((u >> 16) & 1u);
    return (unsigned short)(r >> 16);
}

// 16-byte global -> LDS copy; async (ASYNCcnt-tracked, no VGPR staging) when available.
__device__ inline void copy16_g2l(const unsigned short* g, unsigned short* l) {
#ifdef HAVE_ASYNC_LDS
    __builtin_amdgcn_global_load_async_to_lds_b128((gv4i*)g, (lv4i*)l, 0, 0);
#else
    *reinterpret_cast<uint4*>(l) = *reinterpret_cast<const uint4*>(g);
#endif
}

__device__ inline void wait_async0() {
#if defined(__gfx1250__)
#if __has_builtin(__builtin_amdgcn_s_wait_asynccnt)
    __builtin_amdgcn_s_wait_asynccnt(0);
#else
    asm volatile("s_wait_asynccnt 0x0" ::: "memory");
#endif
#endif
}

__device__ inline v16bf load_frag(const unsigned short* row_base, int c0e) {
    // c0e = 0 (lanes 0-15) or 8 (lanes 16-31); second chunk at +16 elements.
    union { uint4 u[2]; v16bf v; } f;
    f.u[0] = *reinterpret_cast<const uint4*>(row_base + c0e);
    f.u[1] = *reinterpret_cast<const uint4*>(row_base + c0e + 16);
    return f.v;
}

// ---------- kernel 0: zero the stat accumulators ----------
__global__ void zero_stats_kernel(float* p, int n) {
    int i = blockIdx.x * blockDim.x + threadIdx.x;
    if (i < n) p[i] = 0.0f;
}

// ---------- kernel 1: binarize weights (sign * prune-mask) -> bf16 bits ----------
__global__ void binarize_w_kernel(const float* __restrict__ W, unsigned short* __restrict__ Wb) {
    __shared__ float red[256];
    const int row = blockIdx.x;
    const int t = threadIdx.x;
    const float* wr = W + (size_t)row * IN_DIM;
    float vals[16];
    float asum = 0.0f;
#pragma unroll
    for (int i = 0; i < 16; ++i) {
        float w = wr[t + i * 256];
        vals[i] = w;
        asum += fabsf(w);
    }
    red[t] = asum;
    __syncthreads();
    for (int s = 128; s > 0; s >>= 1) {
        if (t < s) red[t] += red[t + s];
        __syncthreads();
    }
    const bool alive = (red[0] != 0.0f);
    unsigned short* wo = Wb + (size_t)row * IN_DIM;
#pragma unroll
    for (int i = 0; i < 16; ++i) {
        float w = vals[i];
        unsigned short b = (w > 0.0f) ? 0x3F80u : ((w < 0.0f) ? 0xBF80u : 0u); // +1/-1/0 in bf16
        if (!alive) b = 0u;
        wo[t + i * 256] = b;
    }
}

// ---------- kernel 2: x f32 -> bf16 (RNE) ----------
__global__ void convert_x_kernel(const float* __restrict__ X, unsigned short* __restrict__ Xb) {
    size_t idx = (size_t)blockIdx.x * blockDim.x + threadIdx.x;
    size_t base = idx * 4;
    float4 v = *reinterpret_cast<const float4*>(X + base);
    uint2 p;
    p.x = (unsigned int)f2bf_rne(v.x) | ((unsigned int)f2bf_rne(v.y) << 16);
    p.y = (unsigned int)f2bf_rne(v.z) | ((unsigned int)f2bf_rne(v.w) << 16);
    *reinterpret_cast<uint2*>(Xb + base) = p;
}

// ---------- kernel 3: WMMA bf16 GEMM + bias + column stats ----------
__global__ __launch_bounds__(256) void gemm_bn_kernel(
    const unsigned short* __restrict__ Xb,   // [B_DIM][IN_DIM] bf16
    const unsigned short* __restrict__ Wb,   // [OUT_DIM][IN_DIM] bf16 (row o = column o of B matrix)
    const float* __restrict__ bias,
    float* __restrict__ out,                 // [B_DIM][OUT_DIM] raw (pre-BN) result
    float* __restrict__ colsum,
    float* __restrict__ colsumsq)
{
    __shared__ __align__(16) unsigned short sA[2][BM * LDSS];
    __shared__ __align__(16) unsigned short sB[2][BN * LDSS];

    const int tid  = threadIdx.x;
    const int lane = tid & 31;
    const int wave = tid >> 5;          // 8 waves
    const int wm   = wave & 1;          // 2 waves along M (64 rows each)
    const int wn   = wave >> 1;         // 4 waves along N (64 cols each)
    const int m0   = blockIdx.y * BM;
    const int n0   = blockIdx.x * BN;

    const int mrow = lane & 15;
    const int c0e  = (lane & 16) ? 8 : 0;  // ISA 16-bit A/B lane K-striping

    v8f acc[4][4] = {};

    // issue all fills for one LDS stage (A: 2 chunks/thread, B: 4 chunks/thread)
    auto stage = [&](int buf, int k0) {
#pragma unroll
        for (int i = 0; i < 2; ++i) {
            int ci = tid + i * 256;
            int row = ci >> 2, c = ci & 3;
            copy16_g2l(&Xb[(size_t)(m0 + row) * IN_DIM + k0 + c * 8],
                       &sA[buf][row * LDSS + c * 8]);
        }
#pragma unroll
        for (int i = 0; i < 4; ++i) {
            int ci = tid + i * 256;
            int row = ci >> 2, c = ci & 3;
            copy16_g2l(&Wb[(size_t)(n0 + row) * IN_DIM + k0 + c * 8],
                       &sB[buf][row * LDSS + c * 8]);
        }
    };

    stage(0, 0);
    int cur = 0;
    for (int k0 = 0; k0 < IN_DIM; k0 += BK) {
#ifdef HAVE_ASYNC_LDS
        wait_async0();                 // own stage-s fills have landed in LDS
#endif
        __syncthreads();               // everyone's fills landed; everyone done reading buf cur^1
        if (k0 + BK < IN_DIM)
            stage(cur ^ 1, k0 + BK);   // overlap next stage's global traffic with WMMA

        const unsigned short* ab = sA[cur];
        const unsigned short* bb = sB[cur];
        v16bf afrag[4], bfrag[4];
#pragma unroll
        for (int i = 0; i < 4; ++i)
            afrag[i] = load_frag(&ab[(wm * 64 + i * 16 + mrow) * LDSS], c0e);
#pragma unroll
        for (int j = 0; j < 4; ++j)
            bfrag[j] = load_frag(&bb[(wn * 64 + j * 16 + mrow) * LDSS], c0e);

#pragma unroll
        for (int i = 0; i < 4; ++i)
#pragma unroll
            for (int j = 0; j < 4; ++j)
                acc[i][j] = __builtin_amdgcn_wmma_f32_16x16x32_bf16(
                    false, afrag[i], false, bfrag[j],
                    (short)0, acc[i][j], false, false);
        cur ^= 1;
    }

    // epilogue: bias add, raw store, per-column sum / sumsq (lane l and l+16 share column)
    const int nlane = lane & 15;
    const int mhalf = (lane & 16) ? 8 : 0;
#pragma unroll
    for (int j = 0; j < 4; ++j) {
        const int n = n0 + wn * 64 + j * 16 + nlane;
        const float bv = bias[n];
        float s = 0.0f, ss = 0.0f;
#pragma unroll
        for (int i = 0; i < 4; ++i) {
            const int mbase = m0 + wm * 64 + i * 16 + mhalf;
#pragma unroll
            for (int r = 0; r < 8; ++r) {
                float v = acc[i][j][r] + bv;
                out[(size_t)(mbase + r) * OUT_DIM + n] = v;
                s  += v;
                ss += v * v;
            }
        }
        s  += __shfl_xor(s, 16, 32);
        ss += __shfl_xor(ss, 16, 32);
        if (lane < 16) {
            atomicAdd(&colsum[n], s);
            atomicAdd(&colsumsq[n], ss);
        }
    }
}

// ---------- kernel 4: fold stats into per-column scale/shift ----------
__global__ void finalize_stats_kernel(const float* __restrict__ colsum,
                                      const float* __restrict__ colsumsq,
                                      const float* __restrict__ gamma,
                                      const float* __restrict__ beta,
                                      float* __restrict__ scale,
                                      float* __restrict__ shift) {
    int n = blockIdx.x * blockDim.x + threadIdx.x;
    if (n < OUT_DIM) {
        const float inv_b = 1.0f / (float)B_DIM;
        float mu  = colsum[n] * inv_b;
        float var = colsumsq[n] * inv_b - mu * mu;
        float sc  = gamma[n] * rsqrtf(var + EPS);
        scale[n] = sc;
        shift[n] = beta[n] - mu * sc;
    }
}

// ---------- kernel 5: in-place normalize ----------
__global__ void normalize_kernel(float* __restrict__ out,
                                 const float* __restrict__ scale,
                                 const float* __restrict__ shift) {
    size_t idx = (size_t)blockIdx.x * blockDim.x + threadIdx.x;
    size_t base = idx * 4;
    int n = (int)(base & (OUT_DIM - 1));
    float4 v = *reinterpret_cast<float4*>(out + base);
    v.x = v.x * scale[n + 0] + shift[n + 0];
    v.y = v.y * scale[n + 1] + shift[n + 1];
    v.z = v.z * scale[n + 2] + shift[n + 2];
    v.w = v.w * scale[n + 3] + shift[n + 3];
    *reinterpret_cast<float4*>(out + base) = v;
}

extern "C" void kernel_launch(void* const* d_in, const int* in_sizes, int n_in,
                              void* d_out, int out_size, void* d_ws, size_t ws_size,
                              hipStream_t stream) {
    const float* x      = (const float*)d_in[0];
    const float* weight = (const float*)d_in[1];
    const float* bias   = (const float*)d_in[2];
    const float* gamma  = (const float*)d_in[3];
    const float* beta   = (const float*)d_in[4];
    float* out = (float*)d_out;

    char* ws = (char*)d_ws;
    unsigned short* Wb = (unsigned short*)ws;                                   // 32 MB
    unsigned short* Xb = (unsigned short*)(ws + (size_t)OUT_DIM * IN_DIM * 2);  // 64 MB
    float* colsum   = (float*)(ws + (size_t)OUT_DIM * IN_DIM * 2
                                  + (size_t)B_DIM * IN_DIM * 2);
    float* colsumsq = colsum + OUT_DIM;
    float* scale    = colsumsq + OUT_DIM;
    float* shift    = scale + OUT_DIM;

    zero_stats_kernel<<<(2 * OUT_DIM + 255) / 256, 256, 0, stream>>>(colsum, 2 * OUT_DIM);
    binarize_w_kernel<<<OUT_DIM, 256, 0, stream>>>(weight, Wb);
    convert_x_kernel<<<(int)(((size_t)B_DIM * IN_DIM / 4) / 256), 256, 0, stream>>>(x, Xb);

    dim3 grid(OUT_DIM / BN, B_DIM / BM);
    gemm_bn_kernel<<<grid, 256, 0, stream>>>(Xb, Wb, bias, out, colsum, colsumsq);

    finalize_stats_kernel<<<OUT_DIM / 256, 256, 0, stream>>>(colsum, colsumsq, gamma, beta, scale, shift);
    normalize_kernel<<<(int)(((size_t)B_DIM * OUT_DIM / 4) / 256), 256, 0, stream>>>(out, scale, shift);
}
